// SparseTopKSelfAttention_8443905704415
// MI455X (gfx1250) — compile-verified, matrix operand order checked by reading, not run on hardware
//
#include <hip/hip_runtime.h>
#include <math.h>

// ---------------------------------------------------------------------------
// Sparse top-k self attention for MI455X (gfx1250, wave32, WMMA)
// B=2, T=2048, D=1024, H=16, HD=64, K_TOP=64
// ---------------------------------------------------------------------------

#define B_ 2
#define T_ 2048
#define D_ 1024
#define H_ 16
#define HD_ 64
#define KTOP_ 64

// LDS Q-panel: 16 rows x 64 bf16, padded to 72 shorts (144 B) per row so the
// ds_load_b128 fragment reads are bank-conflict free (36-bank row step).
#define QPAN_STRIDE 72
#define QPAN_ELEMS  (16 * QPAN_STRIDE)

typedef __attribute__((ext_vector_type(16))) __bf16 v16bf;
typedef __attribute__((ext_vector_type(8)))  float  v8f;

union ABv {
    uint4  u[2];
    v16bf  v;
};

__device__ __forceinline__ unsigned short f2bf(float f) {
    unsigned int u = __float_as_uint(f);
    // round-to-nearest-even bf16
    unsigned int r = (u + 0x7FFFu + ((u >> 16) & 1u)) >> 16;
    return (unsigned short)r;
}

__device__ __forceinline__ v8f wmma_bf16(const ABv& a, const ABv& b, v8f c) {
    // D = A(16x32 bf16) * B(32x16 bf16) + C(16x16 f32)
    return __builtin_amdgcn_wmma_f32_16x16x32_bf16(
        /*neg_a=*/false, a.v, /*neg_b=*/false, b.v,
        /*c_mod=*/(short)0, c, /*reuse_a=*/false, /*reuse_b=*/false);
}

// CDNA5 async copy: global -> LDS, 16 bytes per lane, tracked by ASYNCcnt.
__device__ __forceinline__ void async_ld_b128(unsigned lds_byte_addr, const void* gaddr) {
    asm volatile("global_load_async_to_lds_b128 %0, %1, off"
                 :: "v"(lds_byte_addr), "v"(gaddr)
                 : "memory");
}
__device__ __forceinline__ void wait_async0() {
    asm volatile("s_wait_asynccnt 0" ::: "memory");
}

// ---------------------------------------------------------------------------
// fp32 -> bf16 bulk convert (n must be a multiple of 4; all our sizes are)
// ---------------------------------------------------------------------------
__global__ void __launch_bounds__(256) cvt_bf16_kernel(const float* __restrict__ in,
                                                       unsigned short* __restrict__ out,
                                                       int n4) {
    int i = (blockIdx.x * 256 + threadIdx.x);
    if (i < n4) {
        float4 v = ((const float4*)in)[i];
        ushort4 o;
        o.x = f2bf(v.x); o.y = f2bf(v.y); o.z = f2bf(v.z); o.w = f2bf(v.w);
        ((ushort4*)out)[i] = o;
    }
}

// ---------------------------------------------------------------------------
// Register-blocked 64x64 per-wave WMMA block: 4x4 grid of 16x16 f32 tiles.
// Per K-step of 32: 8 fragment loads (16 x B128) feed 16 WMMAs (1:1 ratio,
// ~32 FLOP/byte).  Both operands are K-major with leading dimension D_
// (compile-time), so row offsets fold into global_load_b128 immediates.
// This variant stays entirely in registers -- the redundant A loads across
// the 8 waves of a block hit L0/WGP$, and avoiding LDS staging keeps VGPR
// pressure under 256 (no s_set_vgpr_msb / v_mov shuffling in the hot loop).
// Lane layout per CDNA5 16-bit A-matrix (16x32):
//   lanes 0-15  : K = k..k+7   (V0-3) and k+16..k+23 (V4-7)
//   lanes 16-31 : K = k+8..k+15 and k+24..k+31
// ---------------------------------------------------------------------------
__device__ __forceinline__ void wmma_block_64x64(
    const unsigned short* __restrict__ Abase,   // A rows at Abase + m*D_
    const unsigned short* __restrict__ Bbase,   // B cols (w rows) at Bbase + n*D_
    int ksteps, v8f acc[4][4]) {
    const int lane = threadIdx.x & 31;
    const int r    = lane & 15;
    const int half = lane >> 4;          // 0 or 1
    const unsigned short* ap = Abase + (size_t)r * D_ + half * 8;
    const unsigned short* bp = Bbase + (size_t)r * D_ + half * 8;
    for (int kk = 0; kk < ksteps; ++kk) {
        ABv a[4], b[4];
#pragma unroll
        for (int i = 0; i < 4; ++i) {
            a[i].u[0] = *(const uint4*)(ap + i * 16 * D_);
            a[i].u[1] = *(const uint4*)(ap + i * 16 * D_ + 16);
            b[i].u[0] = *(const uint4*)(bp + i * 16 * D_);
            b[i].u[1] = *(const uint4*)(bp + i * 16 * D_ + 16);
        }
        __builtin_prefetch(ap + 64, 0, 3);   // global_prefetch_b8 (near)
        __builtin_prefetch(bp + 64, 0, 3);
#pragma unroll
        for (int i = 0; i < 4; ++i)
#pragma unroll
            for (int j = 0; j < 4; ++j)
                acc[i][j] = wmma_bf16(a[i], b[j], acc[i][j]);
        ap += 32;
        bp += 32;
    }
}

// ---------------------------------------------------------------------------
// QKV projection: qkv[m, n] = sum_k x[m,k] * w_qkv[n,k] + b_qkv[n]
// m in [0, B*T), n in [0, 3*D).  Epilogue scatters into head-major
// Q/K (bf16) and V (f32):  [B, H, T, HD].
// 8 waves / block, one 64x64 block / wave.  Wave tiles: 64 x 48 = 3072.
// ---------------------------------------------------------------------------
__global__ void __launch_bounds__(256) qkv_gemm_kernel(
    const unsigned short* __restrict__ xbf,
    const unsigned short* __restrict__ wbf,
    const float* __restrict__ bqkv,
    unsigned short* __restrict__ Qbf,
    unsigned short* __restrict__ Kbf,
    float* __restrict__ Vf) {
    const int NT   = (3 * D_) / 64;               // 48
    const int wave = threadIdx.x >> 5;
    const int tile = blockIdx.x * 8 + wave;
    const int mt = tile / NT;                     // 0..63
    const int nt = tile % NT;                     // 0..47

    v8f acc[4][4] = {};
    wmma_block_64x64(xbf + (size_t)(mt * 64) * D_,
                     wbf + (size_t)(nt * 64) * D_,
                     D_ / 32, acc);

    const int lane  = threadIdx.x & 31;
    const int nloc  = lane & 15;
    const int mbase = (lane >> 4) * 8;

#pragma unroll
    for (int j = 0; j < 4; ++j) {
        const int n     = nt * 64 + j * 16 + nloc;
        const float bias = bqkv[n];
        const int which = n >> 10;        // 0=Q 1=K 2=V
        const int d     = n & (D_ - 1);
        const int h     = d >> 6;
        const int hd    = d & (HD_ - 1);
#pragma unroll
        for (int i = 0; i < 4; ++i) {
#pragma unroll
            for (int r2 = 0; r2 < 8; ++r2) {
                const int m = mt * 64 + i * 16 + mbase + r2;
                const int b = m >> 11;                    // / T_
                const int t = m & (T_ - 1);
                const float val = acc[i][j][r2] + bias;
                const size_t o = (((size_t)(b * H_ + h)) * T_ + t) * HD_ + hd;
                if (which == 0)      Qbf[o] = f2bf(val);
                else if (which == 1) Kbf[o] = f2bf(val);
                else                 Vf[o]  = val;
            }
        }
    }
}

// ---------------------------------------------------------------------------
// Per (b, h, 16-query tile): the shared 16x64 Q tile is staged into LDS once
// per block with CDNA5 async-to-LDS copies (all 16 waves would otherwise
// redundantly load it), then waves WMMA the full 16x2048 score strip into
// LDS, run iterative top-64 (sorted desc, lower index wins ties, like
// jax top_k), softmax over the 64 selected logits, attn @ V gather, and
// write ctx (bf16).  512 threads = 16 waves.
// ---------------------------------------------------------------------------
__global__ void __launch_bounds__(512) attn_topk_kernel(
    const unsigned short* __restrict__ Qbf,
    const unsigned short* __restrict__ Kbf,
    const float* __restrict__ Vf,
    const int* __restrict__ mask,
    float* __restrict__ attn_out,
    int*   __restrict__ idx_out,
    unsigned short* __restrict__ ctx) {
    extern __shared__ float smem[];
    float* sc   = smem;                                   // [16][2048]
    float* topv = sc + 16 * T_;                           // [16][64]
    int*   topi = (int*)(topv + 16 * KTOP_);              // [16][64]
    float* atn  = (float*)(topi + 16 * KTOP_);            // [16][64]
    unsigned short* qpan = (unsigned short*)(atn + 16 * KTOP_);  // [16][72]

    const int wid  = threadIdx.x >> 5;
    const int lane = threadIdx.x & 31;
    const int bh   = blockIdx.x >> 7;           // / (T/16)
    const int qt   = blockIdx.x & 127;
    const int b    = bh >> 4;
    const int h    = bh & (H_ - 1);

    const unsigned short* qbase = Qbf + ((size_t)bh * T_ + qt * 16) * HD_;
    const unsigned short* kbase = Kbf + (size_t)bh * T_ * HD_;

    // ---- async-stage the 16x64 Q tile into LDS (128 x B128 copies) ----
    if (threadIdx.x < 128) {
        const int row = threadIdx.x >> 3;      // 0..15
        const int seg = threadIdx.x & 7;       // 8 x 16B per 128B row
        async_ld_b128((unsigned)(size_t)(qpan + row * QPAN_STRIDE + seg * 8),
                      qbase + (size_t)row * HD_ + seg * 8);
    }
    wait_async0();
    __syncthreads();

    const int r    = lane & 15;
    const int half = lane >> 4;

    // A operands from LDS: Q tile 16x64 -> two 16x32 chunks per wave
    ABv a0, a1;
    {
        const unsigned short* lp = qpan + r * QPAN_STRIDE + half * 8;
        a0.u[0] = *(const uint4*)(lp);
        a0.u[1] = *(const uint4*)(lp + 16);
        a1.u[0] = *(const uint4*)(lp + 32);
        a1.u[1] = *(const uint4*)(lp + 48);
    }

    const float scale = 0.125f;  // 1/sqrt(64)
    const int mbase = (lane >> 4) * 8;

    // scores: wave wid covers key-column tiles wid, wid+16, ...
    for (int ct = wid; ct < T_ / 16; ct += 16) {
        const unsigned short* bp = kbase + (size_t)(ct * 16 + r) * HD_ + half * 8;
        ABv b0, b1;
        b0.u[0] = *(const uint4*)(bp);
        b0.u[1] = *(const uint4*)(bp + 16);
        b1.u[0] = *(const uint4*)(bp + 32);
        b1.u[1] = *(const uint4*)(bp + 48);
        v8f c = {};
        c = wmma_bf16(a0, b0, c);
        c = wmma_bf16(a1, b1, c);

        const int key = ct * 16 + (lane & 15);
        const float madd = (mask[b * T_ + key] == 0) ? -INFINITY : 0.0f;
        for (int r2 = 0; r2 < 8; ++r2)
            sc[(mbase + r2) * T_ + key] = c[r2] * scale + madd;
    }
    __syncthreads();

    // ---- top-64 selection: wave wid owns query row wid ----
    const int row = wid;
    volatile float* srow = sc + row * T_;
    for (int it = 0; it < KTOP_; ++it) {
        float best = -INFINITY;
        int   bidx = 0x7fffffff;
        for (int j = 0; j < T_ / 32; ++j) {
            const int i2 = lane + j * 32;
            const float v = srow[i2];
            if (v > best || (v == best && i2 < bidx)) { best = v; bidx = i2; }
        }
        for (int md = 16; md >= 1; md >>= 1) {
            const float ov = __shfl_xor(best, md, 32);
            const int   oi = __shfl_xor(bidx, md, 32);
            if (ov > best || (ov == best && oi < bidx)) { best = ov; bidx = oi; }
        }
        if (lane == (bidx & 31)) srow[bidx] = -INFINITY;  // remove winner
        if (lane == 0) { topv[row * KTOP_ + it] = best; topi[row * KTOP_ + it] = bidx; }
    }
    __threadfence_block();

    // ---- softmax over the 64 selected logits (first pick is the max) ----
    const float maxv = topv[row * KTOP_];
    const float e0 = __expf(topv[row * KTOP_ + lane]      - maxv);
    const float e1 = __expf(topv[row * KTOP_ + 32 + lane] - maxv);
    float s = e0 + e1;
    for (int md = 16; md >= 1; md >>= 1) s += __shfl_xor(s, md, 32);
    const float inv = 1.0f / s;

    const size_t grow = (size_t)bh * T_ + qt * 16 + row;   // global (b,h,t) row
    attn_out[grow * KTOP_ + lane]      = e0 * inv;
    attn_out[grow * KTOP_ + 32 + lane] = e1 * inv;
    idx_out[grow * KTOP_ + lane]       = topi[row * KTOP_ + lane];
    idx_out[grow * KTOP_ + 32 + lane]  = topi[row * KTOP_ + 32 + lane];
    atn[row * KTOP_ + lane]      = e0 * inv;
    atn[row * KTOP_ + 32 + lane] = e1 * inv;

    // ---- out = attn @ Vsel ; lane covers hd = lane and lane+32 ----
    float acc0 = 0.f, acc1 = 0.f;
    const float* vb = Vf + (size_t)bh * T_ * HD_;
    for (int j = 0; j < KTOP_; ++j) {
        const float w  = atn[row * KTOP_ + j];
        const int   ky = topi[row * KTOP_ + j];
        const float* vr = vb + (size_t)ky * HD_;
        acc0 = fmaf(w, vr[lane], acc0);
        acc1 = fmaf(w, vr[lane + 32], acc1);
    }
    const int t = qt * 16 + row;
    const size_t co = ((size_t)b * T_ + t) * D_ + h * HD_;
    ctx[co + lane]      = f2bf(acc0);
    ctx[co + 32 + lane] = f2bf(acc1);
}

// ---------------------------------------------------------------------------
// Output projection: out[m, n] = sum_k ctx[m,k] * w_out[n,k] + b_out[n]
// 64x64 block per wave, 8 waves / block.  Wave tiles: 64 x 16 = 1024.
// ---------------------------------------------------------------------------
__global__ void __launch_bounds__(256) out_gemm_kernel(
    const unsigned short* __restrict__ ctx,
    const unsigned short* __restrict__ wbf,
    const float* __restrict__ bout,
    float* __restrict__ out) {
    const int NT   = D_ / 64;                     // 16
    const int wave = threadIdx.x >> 5;
    const int tile = blockIdx.x * 8 + wave;
    const int mt = tile / NT;                     // 0..63
    const int nt = tile % NT;                     // 0..15

    v8f acc[4][4] = {};
    wmma_block_64x64(ctx + (size_t)(mt * 64) * D_,
                     wbf + (size_t)(nt * 64) * D_,
                     D_ / 32, acc);

    const int lane  = threadIdx.x & 31;
    const int nloc  = lane & 15;
    const int mbase = (lane >> 4) * 8;
#pragma unroll
    for (int j = 0; j < 4; ++j) {
        const int n = nt * 64 + j * 16 + nloc;
        const float bias = bout[n];
#pragma unroll
        for (int i = 0; i < 4; ++i) {
#pragma unroll
            for (int r2 = 0; r2 < 8; ++r2) {
                const int m = mt * 64 + i * 16 + mbase + r2;
                out[(size_t)m * D_ + n] = acc[i][j][r2] + bias;
            }
        }
    }
}

// ---------------------------------------------------------------------------
// Host-side orchestration
// ---------------------------------------------------------------------------
extern "C" void kernel_launch(void* const* d_in, const int* in_sizes, int n_in,
                              void* d_out, int out_size, void* d_ws, size_t ws_size,
                              hipStream_t stream) {
    (void)in_sizes; (void)n_in; (void)out_size; (void)ws_size;

    const float* x    = (const float*)d_in[0];   // [B,T,D]
    const int*   msk  = (const int*)  d_in[1];   // [B,T]
    const float* wqkv = (const float*)d_in[2];   // [3D,D]
    const float* bqkv = (const float*)d_in[3];   // [3D]
    const float* wout = (const float*)d_in[4];   // [D,D]
    const float* bout = (const float*)d_in[5];   // [D]

    float* out      = (float*)d_out;                               // [B,T,D]
    float* attn_out = out + (size_t)B_ * T_ * D_;                  // [B,H,T,64]
    int*   idx_out  = (int*)(attn_out + (size_t)B_ * H_ * T_ * KTOP_);

    // workspace carve-up
    char* ws = (char*)d_ws;
    unsigned short* xbf  = (unsigned short*)(ws);                       // 8 MB
    unsigned short* wqb  = (unsigned short*)(ws + (8u << 20));          // 6 MB
    unsigned short* wob  = (unsigned short*)(ws + (14u << 20));         // 2 MB
    unsigned short* Qbf  = (unsigned short*)(ws + (16u << 20));         // 8 MB
    unsigned short* Kbf  = (unsigned short*)(ws + (24u << 20));         // 8 MB
    float*          Vf   = (float*)        (ws + (32u << 20));          // 16 MB
    unsigned short* ctx  = (unsigned short*)(ws + (48u << 20));         // 8 MB

    // 1) bf16 conversions
    {
        int n4 = (B_ * T_ * D_) / 4;
        cvt_bf16_kernel<<<(n4 + 255) / 256, 256, 0, stream>>>(x, xbf, n4);
        n4 = (3 * D_ * D_) / 4;
        cvt_bf16_kernel<<<(n4 + 255) / 256, 256, 0, stream>>>(wqkv, wqb, n4);
        n4 = (D_ * D_) / 4;
        cvt_bf16_kernel<<<(n4 + 255) / 256, 256, 0, stream>>>(wout, wob, n4);
    }

    // 2) QKV projection (WMMA, 64x64 register blocks): 3072 wave-tiles
    {
        const int tiles = ((B_ * T_) / 64) * ((3 * D_) / 64);  // 3072
        qkv_gemm_kernel<<<tiles / 8, 256, 0, stream>>>(xbf, wqb, bqkv, Qbf, Kbf, Vf);
    }

    // 3) attention + top-k + softmax + gather (async-staged Q tile)
    {
        const int blocks = B_ * H_ * (T_ / 16);                // 4096
        const size_t smem = (size_t)16 * T_ * sizeof(float)    // scores 128KB
                          + (size_t)16 * KTOP_ * sizeof(float) // topv
                          + (size_t)16 * KTOP_ * sizeof(int)   // topi
                          + (size_t)16 * KTOP_ * sizeof(float) // attn
                          + (size_t)QPAN_ELEMS * sizeof(unsigned short); // Q panel
        attn_topk_kernel<<<blocks, 512, smem, stream>>>(Qbf, Kbf, Vf, msk,
                                                        attn_out, idx_out, ctx);
    }

    // 4) output projection (WMMA, 64x64 register blocks): 1024 wave-tiles
    {
        const int tiles = ((B_ * T_) / 64) * (D_ / 64);        // 1024
        out_gemm_kernel<<<tiles / 8, 256, 0, stream>>>(ctx, wob, bout, out);
    }
}
